// Block_67473936220440
// MI455X (gfx1250) — matrix operand order repaired and checked by reading, not compiled, not run on hardware
//
#include <hip/hip_runtime.h>
#include <hip/hip_bf16.h>
#include <math.h>

// ---------------- problem constants (match reference) ----------------
#define B_    8
#define NX_   2048
#define NY_   2048
#define C_    512
#define H_    4
#define DH_   128            // C/H
#define HID_  2048
#define EPS_  1e-5f

typedef __bf16 bf16;
typedef __attribute__((ext_vector_type(16))) __bf16 v16bf;
typedef __attribute__((ext_vector_type(8)))  __bf16 v8bf;
typedef __attribute__((ext_vector_type(8)))  float  v8f;
typedef __attribute__((ext_vector_type(4)))  unsigned int uv4;
typedef __attribute__((ext_vector_type(8)))  int iv8;
typedef __attribute__((ext_vector_type(4)))  int iv4;

#if defined(__has_builtin)
#if __has_builtin(__builtin_amdgcn_tensor_load_to_lds)
#define USE_TDM 1
#endif
#endif

static __device__ __forceinline__ v8f wmma_bf16(v16bf a, v16bf b, v8f c) {
  // D = A(16x32) * B(32x16) + C, f32 accum
  return __builtin_amdgcn_wmma_f32_16x16x32_bf16(false, a, false, b, (short)0, c,
                                                 false, false);
}

static __device__ __forceinline__ v16bf combine8(v8bf lo, v8bf hi) {
  v16bf r;
#pragma unroll
  for (int i = 0; i < 8; ++i) { r[i] = lo[i]; r[i + 8] = hi[i]; }
  return r;
}

static __device__ __forceinline__ v8f zero8() {
  v8f r;
#pragma unroll
  for (int i = 0; i < 8; ++i) r[i] = 0.0f;
  return r;
}

// ---------------- elementwise casts ----------------
__global__ void cast_bf16_kernel(const float* __restrict__ in, bf16* __restrict__ out, int n) {
  int i = blockIdx.x * blockDim.x + threadIdx.x;
  if (i < n) out[i] = (bf16)in[i];
}

// W[K x N] (row-major, fan_out contiguous) -> WT[N x K] bf16
__global__ void transpose_cast_kernel(const float* __restrict__ W, bf16* __restrict__ WT,
                                      int K, int N) {
  int i = blockIdx.x * blockDim.x + threadIdx.x;
  if (i < K * N) {
    int k = i / N, n = i % N;
    WT[(size_t)n * K + k] = (bf16)W[i];
  }
}

// ---------------- generic bf16 WMMA GEMM: out = act(A @ W + bias) ----------------
// A: [M x K] bf16, WT: [N x K] bf16 (pre-transposed), bias: [N] f32
// Block: 256 thr = 8 waves (4 x 2); block tile 128x128; wave tile 32x64.
template <int RELU, int OF32, int OB16>
__global__ __launch_bounds__(256) void gemm_bf16_kernel(
    const bf16* __restrict__ A, const bf16* __restrict__ WT,
    const float* __restrict__ bias, float* __restrict__ outF,
    bf16* __restrict__ outB, int M, int N, int K) {
  const int lane  = threadIdx.x & 31;
  const int wid   = threadIdx.x >> 5;
  const int waveM = wid >> 1;
  const int waveN = wid & 1;
  const int m0 = blockIdx.y * 128 + waveM * 32;
  const int n0 = blockIdx.x * 128 + waveN * 64;
  const int colL = lane & 15;
  const int kbA = (lane < 16) ? 0 : 8;   // A-frag K offset per half-wave (ISA 7.12.2)
  const int kbB = (lane < 16) ? 0 : 8;   // B-frag K offset per half-wave

  v8f acc[2][4];
#pragma unroll
  for (int i = 0; i < 2; ++i)
#pragma unroll
    for (int j = 0; j < 4; ++j) acc[i][j] = zero8();

  const bf16* a0 = A + (size_t)(m0 + colL) * K;
  const bf16* a1 = A + (size_t)(m0 + 16 + colL) * K;

  for (int kk = 0; kk < K; kk += 32) {
    v16bf af0 = combine8(*(const v8bf*)(a0 + kk + kbA), *(const v8bf*)(a0 + kk + kbA + 16));
    v16bf af1 = combine8(*(const v8bf*)(a1 + kk + kbA), *(const v8bf*)(a1 + kk + kbA + 16));
#pragma unroll
    for (int j = 0; j < 4; ++j) {
      const bf16* bp = WT + (size_t)(n0 + j * 16 + colL) * K + kk + kbB;
      v16bf bf = combine8(*(const v8bf*)bp, *(const v8bf*)(bp + 16));
      acc[0][j] = wmma_bf16(af0, bf, acc[0][j]);
      acc[1][j] = wmma_bf16(af1, bf, acc[1][j]);
    }
  }

  const int rowOff = (lane < 16) ? 0 : 8;  // C-layout: VGPR r -> row r / r+8
#pragma unroll
  for (int i = 0; i < 2; ++i)
#pragma unroll
    for (int j = 0; j < 4; ++j) {
      int n = n0 + j * 16 + colL;
      float bv = bias[n];
#pragma unroll
      for (int r = 0; r < 8; ++r) {
        float v = acc[i][j][r] + bv;
        if (RELU) v = fmaxf(v, 0.0f);
        size_t off = (size_t)(m0 + i * 16 + rowOff + r) * N + n;
        if (OF32) outF[off] = v;
        if (OB16) outB[off] = (bf16)v;
      }
    }
}

// ---------------- fused flash attention (bf16 in, bf16 out) ----------------
// Q,K,V: [B, Nseq, C] bf16 (heads = slices of DH_). One wave = 16 query rows.
// Block = 8 waves = 128 query rows on one (b,h); key tiles of 32.
__global__ __launch_bounds__(256) void flash_attn_kernel(
    const bf16* __restrict__ Qb, const bf16* __restrict__ Kb,
    const bf16* __restrict__ Vb, bf16* __restrict__ Ob,
    int NQ, int NK, float scale) {
  __shared__ __align__(16) bf16 ldsV[32 * 128];      // V tile, keys x d, row-major
  __shared__ __align__(16) bf16 ldsP[8][16 * 32];    // per-wave P transpose buffer

  const int lane = threadIdx.x & 31;
  const int wid  = threadIdx.x >> 5;
  const int bh = blockIdx.y;
  const int b = bh / H_;
  const int h = bh % H_;
  const int q0 = blockIdx.x * 128 + wid * 16;
  const int colL = lane & 15;
  const int halfOff = (lane < 16) ? 0 : 8;  // C-layout half-wave row offset
  const int kbA = (lane < 16) ? 0 : 16;     // A-frag K offset
  const int kbB = (lane < 16) ? 0 : 16;     // B-frag K offset

  // Q fragments: 16x128 as four 16x32 A operands, resident in VGPRs
  v16bf qa[4];
  {
    const bf16* qrow = Qb + (size_t)(b * NQ + q0 + colL) * C_ + h * DH_;
#pragma unroll
    for (int c = 0; c < 4; ++c) {
      const bf16* p = qrow + c * 32 + kbA;
      qa[c] = combine8(*(const v8bf*)p, *(const v8bf*)(p + 8));
    }
  }

  v8f acc[8];
#pragma unroll
  for (int t = 0; t < 8; ++t) acc[t] = zero8();
  float mrun[8], lrun[8];
#pragma unroll
  for (int r = 0; r < 8; ++r) { mrun[r] = -1e30f; lrun[r] = 0.0f; }

  const bf16* kbase = Kb + (size_t)b * NK * C_ + h * DH_;
  const bf16* vbase = Vb + (size_t)b * NK * C_ + h * DH_;
  const unsigned ldsVaddr = (unsigned)(size_t)(&ldsV[0]);  // low 32b of generic = LDS byte offset

  for (int kt = 0; kt < NK; kt += 32) {
    __syncthreads();  // previous tile fully consumed before overwrite
    const bf16* vtile = vbase + (size_t)kt * C_;
#ifdef USE_TDM
    if (wid == 0) {
      // Tensor Data Mover: 32 x 128 bf16 tile, row stride C_ elements -> LDS.
      unsigned long long ga = (unsigned long long)(size_t)vtile;
      uv4 g0 = { 1u,                        // count=1 (valid descriptor)
                 ldsVaddr,                  // lds_addr [63:32]
                 (unsigned)ga,              // global_addr[31:0]
                 ((unsigned)(ga >> 32) & 0x1FFFFFFu) | (2u << 30) };  // addr[56:32] | type=2
      iv8 g1 = { (int)(1u << 16),           // data_size=1 (2 bytes/elem)
                 (int)(128u << 16),         // tensor_dim0[15:0] in [31:16]
                 (int)(32u << 16),          // tensor_dim1[15:0] in [31:16]
                 (int)(128u << 16),         // tile_dim0 = 128
                 (int)32,                   // tile_dim1 = 32, tile_dim2 = 0
                 (int)C_,                   // tensor_dim0_stride[31:0] = 512 elems
                 0, 0 };
      iv4 gz4 = { 0, 0, 0, 0 };
      iv8 gz8 = { 0, 0, 0, 0, 0, 0, 0, 0 };
      __builtin_amdgcn_tensor_load_to_lds(g0, g1, gz4, gz4, gz8, 0);
      __builtin_amdgcn_s_wait_tensorcnt(0);
    }
#else
    for (int c = threadIdx.x; c < 512; c += 256) {   // 512 chunks of 8 bf16
      int row = c >> 4, off = (c & 15) * 8;
      *(v8bf*)&ldsV[row * 128 + off] = *(const v8bf*)(vtile + (size_t)row * C_ + off);
    }
#endif
    __syncthreads();

    // ---- S = Q * K^T for two 16x16 score tiles (keys kt..+15, kt+16..+31)
    v8f s0 = zero8(), s1 = zero8();
    const bf16* k0p = kbase + (size_t)(kt + colL) * C_;
    const bf16* k1p = kbase + (size_t)(kt + 16 + colL) * C_;
#pragma unroll
    for (int c = 0; c < 4; ++c) {
      v16bf kb0 = combine8(*(const v8bf*)(k0p + c * 32 + kbB),
                           *(const v8bf*)(k0p + c * 32 + kbB + 8));
      v16bf kb1 = combine8(*(const v8bf*)(k1p + c * 32 + kbB),
                           *(const v8bf*)(k1p + c * 32 + kbB + 8));
      s0 = wmma_bf16(qa[c], kb0, s0);
      s1 = wmma_bf16(qa[c], kb1, s1);
    }

    // ---- online softmax (rows live in half-wave lane groups of 16)
    float mloc[8];
#pragma unroll
    for (int r = 0; r < 8; ++r) {
      s0[r] *= scale; s1[r] *= scale;
      mloc[r] = fmaxf(s0[r], s1[r]);
    }
#pragma unroll
    for (int off = 8; off >= 1; off >>= 1)
#pragma unroll
      for (int r = 0; r < 8; ++r)
        mloc[r] = fmaxf(mloc[r], __shfl_xor(mloc[r], off, 32));

    float corr[8], lloc[8];
#pragma unroll
    for (int r = 0; r < 8; ++r) {
      float mnew = fmaxf(mrun[r], mloc[r]);
      corr[r] = __expf(mrun[r] - mnew);
      mrun[r] = mnew;
      s0[r] = __expf(s0[r] - mnew);
      s1[r] = __expf(s1[r] - mnew);
      lloc[r] = s0[r] + s1[r];
    }
#pragma unroll
    for (int off = 8; off >= 1; off >>= 1)
#pragma unroll
      for (int r = 0; r < 8; ++r)
        lloc[r] += __shfl_xor(lloc[r], off, 32);
#pragma unroll
    for (int r = 0; r < 8; ++r) lrun[r] = lrun[r] * corr[r] + lloc[r];
#pragma unroll
    for (int t = 0; t < 8; ++t)
#pragma unroll
      for (int r = 0; r < 8; ++r) acc[t][r] *= corr[r];

    // ---- P: C-layout -> A-layout via per-wave LDS (same-wave LDS is in-order)
    bf16* pw = &ldsP[wid][0];
#pragma unroll
    for (int r = 0; r < 8; ++r) {
      pw[(r + halfOff) * 32 + colL]      = (bf16)s0[r];
      pw[(r + halfOff) * 32 + colL + 16] = (bf16)s1[r];
    }
    __builtin_amdgcn_wave_barrier();
    v16bf pa;
    {
      const bf16* pr = pw + colL * 32 + kbA;
      pa = combine8(*(const v8bf*)pr, *(const v8bf*)(pr + 8));
    }

    // ---- O += P(16x32) * V(32x128), V B-operands read from LDS tile
    const int kk0 = (lane < 16) ? 0 : 16;
#pragma unroll
    for (int t = 0; t < 8; ++t) {
      v16bf vbf;
      int d = t * 16 + colL;
#pragma unroll
      for (int j = 0; j < 16; ++j) vbf[j] = ldsV[(kk0 + j) * 128 + d];
      acc[t] = wmma_bf16(pa, vbf, acc[t]);
    }
  }

  // ---- normalize and store bf16 output
  float inv[8];
#pragma unroll
  for (int r = 0; r < 8; ++r) inv[r] = 1.0f / lrun[r];
  bf16* obase = Ob + (size_t)(b * NQ + q0) * C_ + h * DH_;
#pragma unroll
  for (int t = 0; t < 8; ++t)
#pragma unroll
    for (int r = 0; r < 8; ++r)
      obase[(size_t)(r + halfOff) * C_ + t * 16 + colL] = (bf16)(acc[t][r] * inv[r]);
}

// ---------------- fused residual + LayerNorm (wave per row) ----------------
__global__ __launch_bounds__(256) void ln_residual_kernel(
    const float* __restrict__ a, const float* __restrict__ b,
    const float* __restrict__ g, const float* __restrict__ beta,
    float* __restrict__ outF, bf16* __restrict__ outB) {
  const int lane = threadIdx.x & 31;
  const int wid  = threadIdx.x >> 5;
  const size_t row = (size_t)blockIdx.x * 8 + wid;
  const float* ar = a + row * C_;
  const float* br = b + row * C_;
  float v[16];
  float s = 0.0f;
#pragma unroll
  for (int i = 0; i < 16; ++i) {
    int c = lane + i * 32;
    v[i] = ar[c] + br[c];
    s += v[i];
  }
#pragma unroll
  for (int off = 16; off >= 1; off >>= 1) s += __shfl_xor(s, off, 32);
  float mu = s * (1.0f / C_);
  float ss = 0.0f;
#pragma unroll
  for (int i = 0; i < 16; ++i) { float d = v[i] - mu; ss += d * d; }
#pragma unroll
  for (int off = 16; off >= 1; off >>= 1) ss += __shfl_xor(ss, off, 32);
  float rinv = rsqrtf(ss * (1.0f / C_) + EPS_);
#pragma unroll
  for (int i = 0; i < 16; ++i) {
    int c = lane + i * 32;
    float o = (v[i] - mu) * rinv * g[c] + beta[c];
    if (outF) outF[row * C_ + c] = o;
    if (outB) outB[row * C_ + c] = (bf16)o;
  }
}

// ---------------- host orchestration ----------------
extern "C" void kernel_launch(void* const* d_in, const int* in_sizes, int n_in,
                              void* d_out, int out_size, void* d_ws, size_t ws_size,
                              hipStream_t stream) {
  (void)in_sizes; (void)n_in; (void)out_size; (void)ws_size;
  const float* x    = (const float*)d_in[0];
  const float* y    = (const float*)d_in[1];
  const float* saWq = (const float*)d_in[2];   const float* sabq = (const float*)d_in[3];
  const float* saWk = (const float*)d_in[4];   const float* sabk = (const float*)d_in[5];
  const float* saWv = (const float*)d_in[6];   const float* sabv = (const float*)d_in[7];
  const float* saWo = (const float*)d_in[8];   const float* sabo = (const float*)d_in[9];
  const float* caWq = (const float*)d_in[10];  const float* cabq = (const float*)d_in[11];
  const float* caWk = (const float*)d_in[12];  const float* cabk = (const float*)d_in[13];
  const float* caWv = (const float*)d_in[14];  const float* cabv = (const float*)d_in[15];
  const float* caWo = (const float*)d_in[16];  const float* cabo = (const float*)d_in[17];
  const float* lng  = (const float*)d_in[18];  const float* lnb  = (const float*)d_in[19];
  const float* fc1W = (const float*)d_in[20];  const float* fc1b = (const float*)d_in[21];
  const float* fc2W = (const float*)d_in[22];  const float* fc2b = (const float*)d_in[23];

  const int M = B_ * NX_;          // 16384 rows
  char* ws = (char*)d_ws;
  size_t off = 0;
  auto alloc = [&](size_t bytes) -> char* {
    char* p = ws + off;
    off += (bytes + 255) & ~(size_t)255;
    return p;
  };
  bf16* xb   = (bf16*)alloc((size_t)M * C_ * 2);
  bf16* yb   = (bf16*)alloc((size_t)M * C_ * 2);
  bf16* wqT  = (bf16*)alloc((size_t)C_ * C_ * 2);
  bf16* wkT  = (bf16*)alloc((size_t)C_ * C_ * 2);
  bf16* wvT  = (bf16*)alloc((size_t)C_ * C_ * 2);
  bf16* woT  = (bf16*)alloc((size_t)C_ * C_ * 2);
  bf16* cwqT = (bf16*)alloc((size_t)C_ * C_ * 2);
  bf16* cwkT = (bf16*)alloc((size_t)C_ * C_ * 2);
  bf16* cwvT = (bf16*)alloc((size_t)C_ * C_ * 2);
  bf16* cwoT = (bf16*)alloc((size_t)C_ * C_ * 2);
  bf16* fc1T = (bf16*)alloc((size_t)C_ * HID_ * 2);
  bf16* fc2T = (bf16*)alloc((size_t)HID_ * C_ * 2);
  bf16* qb   = (bf16*)alloc((size_t)M * C_ * 2);
  bf16* kb   = (bf16*)alloc((size_t)M * C_ * 2);
  bf16* vb   = (bf16*)alloc((size_t)M * C_ * 2);
  bf16* attb = (bf16*)alloc((size_t)M * C_ * 2);
  float* tmp = (float*)alloc((size_t)M * C_ * 4);
  float* OsaF = (float*)alloc((size_t)M * C_ * 4);
  bf16*  OsaB = (bf16*)alloc((size_t)M * C_ * 2);
  float* OcaF = (float*)alloc((size_t)M * C_ * 4);
  bf16*  OcaB = (bf16*)alloc((size_t)M * C_ * 2);
  bf16*  hb   = (bf16*)alloc((size_t)M * HID_ * 2);

  const int nAct = M * C_;
  cast_bf16_kernel<<<(nAct + 255) / 256, 256, 0, stream>>>(x, xb, nAct);
  cast_bf16_kernel<<<(nAct + 255) / 256, 256, 0, stream>>>(y, yb, nAct);
  const int nW = C_ * C_;
  transpose_cast_kernel<<<(nW + 255) / 256, 256, 0, stream>>>(saWq, wqT, C_, C_);
  transpose_cast_kernel<<<(nW + 255) / 256, 256, 0, stream>>>(saWk, wkT, C_, C_);
  transpose_cast_kernel<<<(nW + 255) / 256, 256, 0, stream>>>(saWv, wvT, C_, C_);
  transpose_cast_kernel<<<(nW + 255) / 256, 256, 0, stream>>>(saWo, woT, C_, C_);
  transpose_cast_kernel<<<(nW + 255) / 256, 256, 0, stream>>>(caWq, cwqT, C_, C_);
  transpose_cast_kernel<<<(nW + 255) / 256, 256, 0, stream>>>(caWk, cwkT, C_, C_);
  transpose_cast_kernel<<<(nW + 255) / 256, 256, 0, stream>>>(caWv, cwvT, C_, C_);
  transpose_cast_kernel<<<(nW + 255) / 256, 256, 0, stream>>>(caWo, cwoT, C_, C_);
  const int nF = C_ * HID_;
  transpose_cast_kernel<<<(nF + 255) / 256, 256, 0, stream>>>(fc1W, fc1T, C_, HID_);
  transpose_cast_kernel<<<(nF + 255) / 256, 256, 0, stream>>>(fc2W, fc2T, HID_, C_);

  dim3 gC(C_ / 128, M / 128);
  dim3 gH(HID_ / 128, M / 128);
  dim3 gA(NX_ / 128, B_ * H_);
  const float sc = 1.0f / sqrtf((float)NX_);   // reference scales by 1/sqrt(n_kv)

  // ---- self attention ----
  gemm_bf16_kernel<0, 0, 1><<<gC, 256, 0, stream>>>(xb, wqT, sabq, nullptr, qb, M, C_, C_);
  gemm_bf16_kernel<0, 0, 1><<<gC, 256, 0, stream>>>(xb, wkT, sabk, nullptr, kb, M, C_, C_);
  gemm_bf16_kernel<0, 0, 1><<<gC, 256, 0, stream>>>(xb, wvT, sabv, nullptr, vb, M, C_, C_);
  flash_attn_kernel<<<gA, 256, 0, stream>>>(qb, kb, vb, attb, NX_, NX_, sc);
  gemm_bf16_kernel<1, 1, 0><<<gC, 256, 0, stream>>>(attb, woT, sabo, tmp, nullptr, M, C_, C_);
  ln_residual_kernel<<<M / 8, 256, 0, stream>>>(tmp, x, lng, lnb, OsaF, OsaB);

  // ---- cross attention ----
  gemm_bf16_kernel<0, 0, 1><<<gC, 256, 0, stream>>>(OsaB, cwqT, cabq, nullptr, qb, M, C_, C_);
  gemm_bf16_kernel<0, 0, 1><<<gC, 256, 0, stream>>>(yb, cwkT, cabk, nullptr, kb, M, C_, C_);
  gemm_bf16_kernel<0, 0, 1><<<gC, 256, 0, stream>>>(yb, cwvT, cabv, nullptr, vb, M, C_, C_);
  flash_attn_kernel<<<gA, 256, 0, stream>>>(qb, kb, vb, attb, NX_, NY_,
                                            1.0f / sqrtf((float)NY_));
  gemm_bf16_kernel<1, 1, 0><<<gC, 256, 0, stream>>>(attb, cwoT, cabo, tmp, nullptr, M, C_, C_);
  ln_residual_kernel<<<M / 8, 256, 0, stream>>>(tmp, OsaF, lng, lnb, OcaF, OcaB);

  // ---- MLP ----
  gemm_bf16_kernel<1, 0, 1><<<gH, 256, 0, stream>>>(OcaB, fc1T, fc1b, nullptr, hb, M, HID_, C_);
  gemm_bf16_kernel<0, 1, 0><<<gC, 256, 0, stream>>>(hb, fc2T, fc2b, tmp, nullptr, M, C_, HID_);
  ln_residual_kernel<<<M / 8, 256, 0, stream>>>(tmp, OcaF, lng, lnb, (float*)d_out, nullptr);
}
